// MRConv2d_72945724555890
// MI455X (gfx1250) — matrix operand order repaired and checked by reading, not compiled, not run on hardware
//
#include <hip/hip_runtime.h>
#include <hip/hip_bf16.h>
#include <math.h>

typedef __attribute__((ext_vector_type(16))) _Float16 v16h;
typedef __attribute__((ext_vector_type(8)))  _Float16 v8h;
typedef __attribute__((ext_vector_type(8)))  float    v8f;

#define BATCH 8
#define NPTS  4096
#define KNN   16
#define CIN   192
#define COUT  192
#define KDIM  (2*CIN)          // 384
#define MROWS (BATCH*NPTS)     // 32768
#define MTILES (MROWS/16)      // 2048
#define BN_EPS 1e-5f

// ---------------------------------------------------------------------------
// Kernel 1: gather + max-relative aggregation -> h [MROWS x 384] f16 row-major
// one block per node, 192 threads (one per channel), neighbor idx in LDS
// ---------------------------------------------------------------------------
__global__ void agg_kernel(const float* __restrict__ x,
                           const int* __restrict__ edge,
                           _Float16* __restrict__ hA) {
    __shared__ int idx[KNN];
    const int node = blockIdx.x;            // 0..MROWS-1
    const int b    = node / NPTS;
    const int c    = threadIdx.x;           // 0..191
    if (c < KNN) idx[c] = edge[node * KNN + c];
    __syncthreads();

    const float xi = x[(size_t)node * CIN + c];
    float m = -3.402823466e38f;
    const float* xb = x + (size_t)b * NPTS * CIN;
#pragma unroll
    for (int k = 0; k < KNN; ++k) {
        float v = xb[(size_t)idx[k] * CIN + c] - xi;
        m = fmaxf(m, v);
    }
    hA[(size_t)node * KDIM + c]       = (_Float16)xi;
    hA[(size_t)node * KDIM + CIN + c] = (_Float16)m;
}

// ---------------------------------------------------------------------------
// Kernel 2: pack W [384 x 192] f32 row-major -> Wt [192 x 384] f16 row-major
// ---------------------------------------------------------------------------
__global__ void packW_kernel(const float* __restrict__ W,
                             _Float16* __restrict__ Wt) {
    int t = blockIdx.x * blockDim.x + threadIdx.x;   // 0 .. 192*384-1
    if (t >= COUT * KDIM) return;
    int n = t / KDIM;
    int k = t % KDIM;
    Wt[(size_t)n * KDIM + k] = (_Float16)W[(size_t)k * COUT + n];
}

// ---------------------------------------------------------------------------
// Kernel 3: GEMM y = h @ W via v_wmma_f32_16x16x32_f16 (bias skipped: BN
// mean-subtraction cancels it exactly).
//
// Each wave computes a 32(M) x 64(N) output tile: 2 M-tiles x 4 N-tiles.
// Per k-step: 2 A-frag + 4 B-frag loads (12 x b128) feed 8 WMMAs, so B
// traffic is amortized over two A tiles and the scheduler has 6 independent
// fragment sets to overlap with MMA issue.
//
// A frag (16x32 f16, documented layout): lane<16 holds row=lane, K = k0+{0..7}
// and k0+{16..23}; lane>=16 holds row=lane-16, K = k0+{8..15}, k0+{24..31}.
// B frag (32x16 f16): same K layout as A: lane<16 -> col=lane, K=k0+{0..7},
// k0+{16..23}; lane>=16 -> col=lane-16, K=k0+{8..15}, k0+{24..31}. Row-major h / Wt => contiguous 16B loads.
// ---------------------------------------------------------------------------
__global__ void gemm_kernel(const _Float16* __restrict__ hA,
                            const _Float16* __restrict__ Wt,
                            float* __restrict__ y) {
    const int lane   = threadIdx.x & 31;
    const int wave   = threadIdx.x >> 5;
    const int mpair  = blockIdx.x * 8 + wave;       // 0..1023 (32-row tiles)
    const int mtile0 = mpair * 2;                   // first 16-row tile
    const int nt0    = blockIdx.y * 4;              // 0,4,8 -> 12 N tiles

    const int arow0 = mtile0 * 16 + (lane & 15);
    const int alo   = (lane < 16) ? 0 : 8;          // first  8 K of k-step
    const int ahi   = (lane < 16) ? 16 : 24;        // second 8 K of k-step
    const int bcol0 = (lane & 15);
    const int bkoff = (lane < 16) ? 0 : 8;

    const _Float16* arow_base0 = hA + (size_t)arow0 * KDIM;
    const _Float16* arow_base1 = arow_base0 + (size_t)16 * KDIM;

    v8f acc[2][4];
#pragma unroll
    for (int m = 0; m < 2; ++m)
#pragma unroll
        for (int j = 0; j < 4; ++j) acc[m][j] = (v8f){};

#pragma unroll
    for (int kt = 0; kt < KDIM / 32; ++kt) {        // 12 k-steps
        const int k0 = kt * 32;

        // ---- issue ALL fragment loads for this k-step first ----
        v8h a0lo = *(const v8h*)(arow_base0 + k0 + alo);
        v8h a0hi = *(const v8h*)(arow_base0 + k0 + ahi);
        v8h a1lo = *(const v8h*)(arow_base1 + k0 + alo);
        v8h a1hi = *(const v8h*)(arow_base1 + k0 + ahi);
        v8h blo[4], bhi[4];
#pragma unroll
        for (int j = 0; j < 4; ++j) {
            const _Float16* bbase =
                Wt + (size_t)((nt0 + j) * 16 + bcol0) * KDIM + k0 + bkoff;
            blo[j] = *(const v8h*)(bbase);
            bhi[j] = *(const v8h*)(bbase + 16);
        }

        v16h a0, a1;
#pragma unroll
        for (int i = 0; i < 8; ++i) {
            a0[i] = a0lo[i]; a0[8 + i] = a0hi[i];
            a1[i] = a1lo[i]; a1[8 + i] = a1hi[i];
        }

        // ---- 8 WMMAs ----
#pragma unroll
        for (int j = 0; j < 4; ++j) {
            v16h bf;
#pragma unroll
            for (int i = 0; i < 8; ++i) { bf[i] = blo[j][i]; bf[8 + i] = bhi[j][i]; }
            acc[0][j] = __builtin_amdgcn_wmma_f32_16x16x32_f16(
                false, a0, false, bf, (short)0, acc[0][j], false, false);
            acc[1][j] = __builtin_amdgcn_wmma_f32_16x16x32_f16(
                false, a1, false, bf, (short)0, acc[1][j], false, false);
        }
    }

    // ---- store C/D: VGPR i: lanes 0-15 -> M=i, N=lane; lanes 16-31 -> M=8+i
    const int cbase = lane & 15;
#pragma unroll
    for (int m = 0; m < 2; ++m) {
        const int rbase = (mtile0 + m) * 16 + ((lane < 16) ? 0 : 8);
#pragma unroll
        for (int j = 0; j < 4; ++j) {
            const int col = (nt0 + j) * 16 + cbase;
#pragma unroll
            for (int i = 0; i < 8; ++i) {
                y[(size_t)(rbase + i) * COUT + col] = acc[m][j][i];
            }
        }
    }
}

// ---------------------------------------------------------------------------
// Kernel 4a: zero the sum/sumsq accumulators
// ---------------------------------------------------------------------------
__global__ void zero_kernel(float* __restrict__ sums) {
    int t = blockIdx.x * blockDim.x + threadIdx.x;
    if (t < 2 * COUT) sums[t] = 0.0f;
}

// ---------------------------------------------------------------------------
// Kernel 4b: per-channel sum / sumsq partial reduction (coarse atomics)
// 256 blocks x 192 threads; each block covers 128 rows.
// ---------------------------------------------------------------------------
__global__ void stats_kernel(const float* __restrict__ y,
                             float* __restrict__ sums) {
    const int c  = threadIdx.x;              // 0..191
    const int r0 = blockIdx.x * (MROWS / 256);
    float s = 0.0f, s2 = 0.0f;
#pragma unroll 4
    for (int r = 0; r < MROWS / 256; ++r) {
        float v = y[(size_t)(r0 + r) * COUT + c];
        s  += v;
        s2 += v * v;
    }
    atomicAdd(&sums[c], s);
    atomicAdd(&sums[COUT + c], s2);
}

// ---------------------------------------------------------------------------
// Kernel 5: batchnorm (batch stats, biased var) + exact-erf GELU
// ---------------------------------------------------------------------------
__global__ void finalize_kernel(const float* __restrict__ y,
                                const float* __restrict__ sums,
                                const float* __restrict__ gamma,
                                const float* __restrict__ beta,
                                float* __restrict__ out) {
    size_t i = (size_t)blockIdx.x * blockDim.x + threadIdx.x;
    if (i >= (size_t)MROWS * COUT) return;
    const int c = (int)(i % COUT);
    const float inv_m = 1.0f / (float)MROWS;
    const float mean  = sums[c] * inv_m;
    const float var   = sums[COUT + c] * inv_m - mean * mean;
    float v = (y[i] - mean) * rsqrtf(var + BN_EPS) * gamma[c] + beta[c];
    out[i] = 0.5f * v * (1.0f + erff(v * 0.70710678118654752f));
}

// ---------------------------------------------------------------------------
// launch
// ---------------------------------------------------------------------------
extern "C" void kernel_launch(void* const* d_in, const int* in_sizes, int n_in,
                              void* d_out, int out_size, void* d_ws, size_t ws_size,
                              hipStream_t stream) {
    // setup_inputs order: x, W, b, gamma, beta, edge_index
    const float* x     = (const float*)d_in[0];
    const float* W     = (const float*)d_in[1];
    // d_in[2] = bias: mathematically cancelled by BN mean subtraction.
    const float* gamma = (const float*)d_in[3];
    const float* beta  = (const float*)d_in[4];
    const int*   edge  = (const int*)d_in[5];
    float* out = (float*)d_out;

    // workspace layout (256B aligned)
    char* ws = (char*)d_ws;
    size_t off = 0;
    _Float16* hA = (_Float16*)(ws + off); off += (size_t)MROWS * KDIM * sizeof(_Float16);
    off = (off + 255) & ~(size_t)255;
    _Float16* Wt = (_Float16*)(ws + off); off += (size_t)COUT * KDIM * sizeof(_Float16);
    off = (off + 255) & ~(size_t)255;
    float* y = (float*)(ws + off);        off += (size_t)MROWS * COUT * sizeof(float);
    off = (off + 255) & ~(size_t)255;
    float* sums = (float*)(ws + off);     off += 2 * COUT * sizeof(float);

    // 1) gather + max-relative agg -> h (f16)
    agg_kernel<<<MROWS, CIN, 0, stream>>>(x, edge, hA);
    // 2) pack W -> Wt (f16, N-major)
    packW_kernel<<<(COUT * KDIM + 255) / 256, 256, 0, stream>>>(W, Wt);
    // 3) WMMA GEMM: 1024 32-row tiles (8 waves/block), 3 groups of 4 N tiles
    dim3 ggrid(MTILES / 2 / 8, 3);
    gemm_kernel<<<ggrid, 256, 0, stream>>>(hA, Wt, y);
    // 4) BN batch statistics
    zero_kernel<<<(2 * COUT + 255) / 256, 256, 0, stream>>>(sums);
    stats_kernel<<<256, COUT, 0, stream>>>(y, sums);
    // 5) normalize + GELU
    size_t total = (size_t)MROWS * COUT;
    finalize_kernel<<<(unsigned)((total + 255) / 256), 256, 0, stream>>>(
        y, sums, gamma, beta, out);
}
